// PCLAD_C_83829171683703
// MI455X (gfx1250) — matrix operand order repaired and checked by reading, not compiled, not run on hardware
//
#include <hip/hip_runtime.h>
#include <hip/hip_bf16.h>
#include <math.h>

// ---------------------------------------------------------------------------
// CDNA5 (gfx1250) wave32 WMMA types
// ---------------------------------------------------------------------------
typedef __attribute__((ext_vector_type(16))) __bf16 v16bf;
typedef __attribute__((ext_vector_type(8)))  float  v8f;

#define WMMA_BF16(a, b, c) \
    __builtin_amdgcn_wmma_f32_16x16x32_bf16(false, (a), false, (b), (short)0, (c), false, false)

union BFV {
    v16bf v;
    unsigned short s[16];
    uint4 q[2];
};

// round-to-nearest-even f32 -> bf16 bits
static __device__ inline unsigned short f2bf(float f) {
    unsigned int u = __float_as_uint(f);
    unsigned int r = u + 0x7FFFu + ((u >> 16) & 1u);
    return (unsigned short)(r >> 16);
}

static __device__ inline float wredsum(float x) {
    for (int o = 16; o > 0; o >>= 1) x += __shfl_xor(x, o, 32);
    return x;
}

static __device__ inline float geluf(float x) {
    return 0.5f * x * (1.0f + erff(x * 0.70710678118654752440f));
}

// ---------------------------------------------------------------------------
// f32 -> bf16 bulk conversion (adjacency, node features)
// ---------------------------------------------------------------------------
__global__ void k_conv_bf16(const float* __restrict__ src, unsigned short* __restrict__ dst, int n) {
    int i = blockIdx.x * blockDim.x + threadIdx.x;
    int st = gridDim.x * blockDim.x;
    for (; i < n; i += st) dst[i] = f2bf(src[i]);
}

__global__ void k_zero_u16(unsigned short* __restrict__ dst, int n) {
    int i = blockIdx.x * blockDim.x + threadIdx.x;
    if (i < n) dst[i] = 0;
}

// ---------------------------------------------------------------------------
// Pack f32 W[Krows,Nc] into bf16 WMMA-B layout, zero-padded to K rows.
// Out index o = ((kblk*ntN) + tn)*512 + lane*16 + j  maps to
//   col = tn*16 + (lane&15),  k = kblk*32 + (lane>>4)*16 + j
// so each lane of a wave later loads its 16 contiguous bf16 with two uint4s.
// ---------------------------------------------------------------------------
__global__ void k_pack_w(const float* __restrict__ W, unsigned short* __restrict__ Wp,
                         int K, int Nc, int Krows) {
    int n = K * Nc;
    int ntN = Nc >> 4;
    int o = blockIdx.x * blockDim.x + threadIdx.x;
    if (o >= n) return;
    int j = o & 15;
    int lane = (o >> 4) & 31;
    int tnk = o >> 9;
    int tn = tnk % ntN;
    int kblk = tnk / ntN;
    int col = tn * 16 + (lane & 15);
    int k = kblk * 32 + (lane >> 4) * 16 + j;
    Wp[o] = (k < Krows) ? f2bf(W[(size_t)k * Nc + col]) : (unsigned short)0;
}

// ---------------------------------------------------------------------------
// phases0 = atan2(sum x*sin(idx), sum x*cos(idx)); one wave per row (D=256)
// ---------------------------------------------------------------------------
__global__ void k_phase_init(const float* __restrict__ x, float* __restrict__ phases, int M) {
    int row = blockIdx.x * 8 + (threadIdx.x >> 5);
    int lane = threadIdx.x & 31;
    if (row >= M) return;
    const float* xr = x + (size_t)row * 256;
    float ss = 0.f, sc = 0.f;
    for (int i = 0; i < 8; i++) {
        int j = lane + 32 * i;
        float v = xr[j];
        float fj = (float)j;
        ss += v * sinf(fj);
        sc += v * cosf(fj);
    }
    ss = wredsum(ss);
    sc = wredsum(sc);
    if (lane == 0) phases[row] = atan2f(ss, sc);
}

// ---------------------------------------------------------------------------
// WMMA GEMM: C[M,Nc] = bf16 A[M,K] @ packed-bf16 Wp + bias, f32 accumulate.
// One 16x16 output tile per wave; 4 waves per block. K % 32 == 0, Nc % 16 == 0.
// Both operands are two uint4 loads per WMMA.
// ---------------------------------------------------------------------------
__global__ __launch_bounds__(128)
void k_gemm_bias(const unsigned short* __restrict__ A,
                 const unsigned short* __restrict__ Wp,
                 const float* __restrict__ bias,
                 float* __restrict__ C,
                 int M, int K, int Nc) {
    int wave = threadIdx.x >> 5;
    int lane = threadIdx.x & 31;
    int ntN = Nc >> 4;
    int tile = blockIdx.x * 4 + wave;
    if (tile >= (M >> 4) * ntN) return;          // wave-uniform: EXEC stays full
    int tm = tile / ntN, tn = tile % ntN;
    int r = lane & 15, half = lane >> 4;
    const unsigned short* arow = A + (size_t)(tm * 16 + r) * K;
    const unsigned short* wlane = Wp + (size_t)tn * 512 + lane * 16;
    v8f acc = {};
#pragma unroll 2
    for (int k0 = 0; k0 < K; k0 += 32) {
        BFV a;
        a.q[0] = *reinterpret_cast<const uint4*>(arow + k0 + half * 16);
        a.q[1] = *reinterpret_cast<const uint4*>(arow + k0 + half * 16 + 8);
        BFV b;
        const unsigned short* wp = wlane + (size_t)(k0 >> 5) * ntN * 512;
        b.q[0] = *reinterpret_cast<const uint4*>(wp);
        b.q[1] = *reinterpret_cast<const uint4*>(wp + 8);
        acc = WMMA_BF16(a.v, b.v, acc);
    }
    int rowb = half * 8;
    int colg = tn * 16 + r;
    float bv = bias[colg];
#pragma unroll
    for (int v = 0; v < 8; v++)
        C[(size_t)(tm * 16 + rowb + v) * Nc + colg] = acc[v] + bv;
}

// ---------------------------------------------------------------------------
// LayerNorm over width 256 + activation, bf16 output; one wave per row
// ---------------------------------------------------------------------------
__global__ void k_ln_act(const float* __restrict__ X, const float* __restrict__ g,
                         const float* __restrict__ bb, unsigned short* __restrict__ O,
                         int M, int mode) {
    int row = blockIdx.x * 8 + (threadIdx.x >> 5);
    int lane = threadIdx.x & 31;
    if (row >= M) return;
    const float* xr = X + (size_t)row * 256;
    float v[8], s = 0.f;
    for (int i = 0; i < 8; i++) { v[i] = xr[lane + 32 * i]; s += v[i]; }
    s = wredsum(s);
    float mean = s * (1.f / 256.f);
    float vs = 0.f;
    for (int i = 0; i < 8; i++) { float d = v[i] - mean; vs += d * d; }
    vs = wredsum(vs);
    float inv = rsqrtf(vs * (1.f / 256.f) + 1e-5f);
    for (int i = 0; i < 8; i++) {
        int j = lane + 32 * i;
        float y = (v[i] - mean) * inv * g[j] + bb[j];
        y = mode ? geluf(y) : tanhf(y);
        O[(size_t)row * 256 + j] = f2bf(y);
    }
}

// in-place elementwise activation (0=tanh, 1=gelu)
__global__ void k_ew_act(float* __restrict__ x, int n, int mode) {
    int i = blockIdx.x * blockDim.x + threadIdx.x;
    int st = gridDim.x * blockDim.x;
    for (; i < n; i += st) {
        float v = x[i];
        x[i] = mode ? geluf(v) : tanhf(v);
    }
}

// ---------------------------------------------------------------------------
// GEMV width 128: out[row] = act(dot(H[row], w) + b); one wave per row.
// mode 1 -> sigmoid. Optional cs_out[row] = out * clip(gc, 0.1, 2.0).
// ---------------------------------------------------------------------------
__global__ void k_gemv128(const float* __restrict__ H, const float* __restrict__ w,
                          const float* __restrict__ bptr, float* __restrict__ out,
                          int M, int mode, float* __restrict__ cs_out,
                          const float* __restrict__ gc_ptr) {
    int row = blockIdx.x * 8 + (threadIdx.x >> 5);
    int lane = threadIdx.x & 31;
    if (row >= M) return;
    const float* hr = H + (size_t)row * 128;
    float s = 0.f;
    for (int i = 0; i < 4; i++) { int j = lane + 32 * i; s += hr[j] * w[j]; }
    s = wredsum(s);
    s += bptr[0];
    if (mode) s = 1.f / (1.f + expf(-s));
    if (lane == 0) {
        out[row] = s;
        if (cs_out) {
            float gcv = fminf(fmaxf(gc_ptr[0], 0.1f), 2.0f);
            cs_out[row] = s * gcv;
        }
    }
}

// ---------------------------------------------------------------------------
// Kuramoto: materialize cos/sin(phases) as bf16
// ---------------------------------------------------------------------------
__global__ void k_prep(const float* __restrict__ ph, unsigned short* __restrict__ pc,
                       unsigned short* __restrict__ ps, int n) {
    int i = blockIdx.x * blockDim.x + threadIdx.x;
    if (i < n) {
        float p = ph[i];
        pc[i] = f2bf(cosf(p));
        ps[i] = f2bf(sinf(p));
    }
}

// ---------------------------------------------------------------------------
// Kuramoto step: one block per 16-row tile; 4 waves split K=8192.
// B matrix = [cosφ | sinφ | 0...], so D[:,0]=ncos, D[:,1]=nsin.
// Per-lane B source pointer selected once (pc / ps / shared zero page), so the
// inner loop is branch-free: 4x global_load_b128 + 1 WMMA per iteration.
// LDS-reduce wave partials, then update the 16 phases in place.
// A (bf16, 128 MB) is resident in the 192 MB L2 after the first sweep.
// ---------------------------------------------------------------------------
__global__ __launch_bounds__(128)
void k_step(const unsigned short* __restrict__ Abf,
            const unsigned short* __restrict__ pcb,
            const unsigned short* __restrict__ psb,
            const unsigned short* __restrict__ zpage,
            const float* __restrict__ freq,
            const float* __restrict__ cs,
            float* __restrict__ phases) {
    __shared__ float lnc[4][16];
    __shared__ float lns[4][16];
    const int N = 8192;
    int wave = threadIdx.x >> 5;
    int lane = threadIdx.x & 31;
    int tm = blockIdx.x;
    int r = lane & 15, half = lane >> 4;
    const unsigned short* arow = Abf + (size_t)(tm * 16 + r) * N;
    const unsigned short* bsrc = (r == 0) ? pcb : (r == 1) ? psb : zpage;
    int kstart = wave * 2048;
    v8f acc = {};
#pragma unroll 8
    for (int k0 = kstart; k0 < kstart + 2048; k0 += 32) {
        BFV a;
        a.q[0] = *reinterpret_cast<const uint4*>(arow + k0 + half * 16);
        a.q[1] = *reinterpret_cast<const uint4*>(arow + k0 + half * 16 + 8);
        BFV b;
        int kb = k0 + half * 16;
        b.q[0] = *reinterpret_cast<const uint4*>(bsrc + kb);
        b.q[1] = *reinterpret_cast<const uint4*>(bsrc + kb + 8);
        acc = WMMA_BF16(a.v, b.v, acc);
    }
    int rowb = half * 8;
    if (r == 0) {
#pragma unroll
        for (int v = 0; v < 8; v++) lnc[wave][rowb + v] = acc[v];
    } else if (r == 1) {
#pragma unroll
        for (int v = 0; v < 8; v++) lns[wave][rowb + v] = acc[v];
    }
    __syncthreads();
    if (threadIdx.x < 16) {
        int rr = threadIdx.x;
        float nc = lnc[0][rr] + lnc[1][rr] + lnc[2][rr] + lnc[3][rr];
        float ns = lns[0][rr] + lns[1][rr] + lns[2][rr] + lns[3][rr];
        int row = tm * 16 + rr;
        float ph = phases[row];
        float ce = atan2f(ns, nc + 1e-8f) - ph;
        float dphi = freq[row] + cs[row] * sinf(ce);
        const float TWO_PI = 6.28318530717958647692f;
        float np_ = fmodf(ph + 0.01f * dphi, TWO_PI);
        if (np_ < 0.f) np_ += TWO_PI;
        phases[row] = np_;
    }
}

// ---------------------------------------------------------------------------
// sync_input: f32 [8192,260] to d_out, bf16 padded [8192,288] for the GEMM
// ---------------------------------------------------------------------------
__global__ void k_build_sync(const float* __restrict__ x, const float* __restrict__ phases,
                             float* __restrict__ syncf, unsigned short* __restrict__ syncbf) {
    int row = blockIdx.x;
    int t = threadIdx.x;
    float val = 0.f;
    bool wf = false;
    if (t < 256) {
        val = x[(size_t)row * 256 + t];
        wf = true;
    } else if (t < 260) {
        float p = phases[row];
        if (t == 256) val = cosf(p);
        else if (t == 257) val = sinf(p);
        else if (t == 258) val = cosf(2.f * p);
        else val = sinf(2.f * p);
        wf = true;
    }
    if (t < 288) syncbf[(size_t)row * 288 + t] = wf ? f2bf(val) : (unsigned short)0;
    if (wf) syncf[(size_t)row * 260 + t] = val;
}

// ---------------------------------------------------------------------------
// Final stats: order parameter, std(ddof=1) -> coherence; copy phases out
// ---------------------------------------------------------------------------
__global__ __launch_bounds__(1024)
void k_stats(const float* __restrict__ phases, float* __restrict__ out) {
    __shared__ float sc_[32], ss_[32], sp_[32], sq_[32];
    int t = threadIdx.x;
    float c = 0.f, s = 0.f, p = 0.f, q = 0.f;
    for (int i = 0; i < 8; i++) {
        int idx = t + 1024 * i;
        float ph = phases[idx];
        c += cosf(ph); s += sinf(ph); p += ph; q += ph * ph;
        out[8193 + idx] = ph;  // phases output
    }
    c = wredsum(c); s = wredsum(s); p = wredsum(p); q = wredsum(q);
    int wave = t >> 5, lane = t & 31;
    if (lane == 0) { sc_[wave] = c; ss_[wave] = s; sp_[wave] = p; sq_[wave] = q; }
    __syncthreads();
    if (t == 0) {
        float C = 0.f, S = 0.f, P = 0.f, Q = 0.f;
        for (int i = 0; i < 32; i++) { C += sc_[i]; S += ss_[i]; P += sp_[i]; Q += sq_[i]; }
        float mc = C / 8192.f, ms = S / 8192.f;
        out[8192] = sqrtf(mc * mc + ms * ms);                       // order_parameter
        float var = (Q - P * P / 8192.f) / 8191.f;                  // ddof=1
        out[2163649] = 1.f / (1.f + sqrtf(fmaxf(var, 0.f)));        // phase_coherence
    }
}

// ---------------------------------------------------------------------------
extern "C" void kernel_launch(void* const* d_in, const int* in_sizes, int n_in,
                              void* d_out, int out_size, void* d_ws, size_t ws_size,
                              hipStream_t stream) {
    (void)in_sizes; (void)n_in; (void)out_size; (void)ws_size;
    const float* x    = (const float*)d_in[0];
    const float* adj  = (const float*)d_in[1];
    const float* fb1  = (const float*)d_in[3];
    const float* flng = (const float*)d_in[4];
    const float* flnb = (const float*)d_in[5];
    const float* fb2  = (const float*)d_in[7];
    const float* fw3  = (const float*)d_in[8];
    const float* fb3  = (const float*)d_in[9];
    const float* cb1  = (const float*)d_in[11];
    const float* cw2  = (const float*)d_in[12];
    const float* cb2  = (const float*)d_in[13];
    const float* sb1  = (const float*)d_in[15];
    const float* slng = (const float*)d_in[16];
    const float* slnb = (const float*)d_in[17];
    const float* sb2  = (const float*)d_in[19];
    const float* sw3  = (const float*)d_in[20];
    const float* sb3  = (const float*)d_in[21];
    const float* gc   = (const float*)d_in[22];

    float* out = (float*)d_out;
    char* ws = (char*)d_ws;

    unsigned short* ABF    = (unsigned short*)(ws + 0);              // 128 MB
    unsigned short* XBF    = (unsigned short*)(ws + 134217728ull);   // 4 MB
    unsigned short* SYNCBF = (unsigned short*)(ws + 138412032ull);   // 4.5 MB
    unsigned short* H1BF   = (unsigned short*)(ws + 143130624ull);   // 4 MB
    float*          T1     = (float*)(ws + 147324928ull);            // 8 MB
    float*          T2     = (float*)(ws + 155713536ull);            // 4 MB
    unsigned short* FW1P   = (unsigned short*)(ws + 159907840ull);   // packed weights
    unsigned short* FW2P   = (unsigned short*)(ws + 160038912ull);
    unsigned short* CW1P   = (unsigned short*)(ws + 160104448ull);
    unsigned short* SW1P   = (unsigned short*)(ws + 160169984ull);
    unsigned short* SW2P   = (unsigned short*)(ws + 160317440ull);
    float*          PHASES = (float*)(ws + 160382976ull);
    float*          CS     = (float*)(ws + 160415744ull);
    unsigned short* PCB    = (unsigned short*)(ws + 160448512ull);
    unsigned short* PSB    = (unsigned short*)(ws + 160464896ull);
    unsigned short* ZPAGE  = (unsigned short*)(ws + 160481280ull);   // 16 KB zeros

    float* FREQ = out + 16385;   // output offsets: desync 0, op 8192, phases 8193,
    float* COUP = out + 24577;   // freq 16385, coupling 24577, sync 32769, coh 2163649
    float* SYNCF = out + 32769;

    // --- one-time conversions + weight packing into WMMA-B layout ---
    k_conv_bf16<<<8192, 256, 0, stream>>>(adj, ABF, 8192 * 8192);
    k_conv_bf16<<<2048, 256, 0, stream>>>(x, XBF, 8192 * 256);
    k_zero_u16<<<32, 256, 0, stream>>>(ZPAGE, 8192);
    k_pack_w<<<256, 256, 0, stream>>>((const float*)d_in[2], FW1P, 256, 256, 256);
    k_pack_w<<<128, 256, 0, stream>>>((const float*)d_in[6], FW2P, 256, 128, 256);
    k_pack_w<<<128, 256, 0, stream>>>((const float*)d_in[10], CW1P, 256, 128, 256);
    k_pack_w<<<288, 256, 0, stream>>>((const float*)d_in[14], SW1P, 288, 256, 260);
    k_pack_w<<<128, 256, 0, stream>>>((const float*)d_in[18], SW2P, 256, 128, 256);

    // --- phase init ---
    k_phase_init<<<1024, 256, 0, stream>>>(x, PHASES, 8192);

    // --- frequency network: Linear->LN->Tanh->Linear->Tanh->Linear ---
    k_gemm_bias<<<2048, 128, 0, stream>>>(XBF, FW1P, fb1, T1, 8192, 256, 256);
    k_ln_act<<<1024, 256, 0, stream>>>(T1, flng, flnb, H1BF, 8192, /*tanh*/0);
    k_gemm_bias<<<1024, 128, 0, stream>>>(H1BF, FW2P, fb2, T2, 8192, 256, 128);
    k_ew_act<<<1024, 256, 0, stream>>>(T2, 8192 * 128, /*tanh*/0);
    k_gemv128<<<1024, 256, 0, stream>>>(T2, fw3, fb3, FREQ, 8192, 0, nullptr, nullptr);

    // --- coupling network: Linear->GELU->Linear->Sigmoid ---
    k_gemm_bias<<<1024, 128, 0, stream>>>(XBF, CW1P, cb1, T2, 8192, 256, 128);
    k_ew_act<<<1024, 256, 0, stream>>>(T2, 8192 * 128, /*gelu*/1);
    k_gemv128<<<1024, 256, 0, stream>>>(T2, cw2, cb2, COUP, 8192, 1, CS, gc);

    // --- Kuramoto integration: 60 steps, A resident in L2 as bf16 ---
    for (int step = 0; step < 60; step++) {
        k_prep<<<32, 256, 0, stream>>>(PHASES, PCB, PSB, 8192);
        k_step<<<512, 128, 0, stream>>>(ABF, PCB, PSB, ZPAGE, FREQ, CS, PHASES);
    }

    // --- sync analyzer ---
    k_build_sync<<<8192, 288, 0, stream>>>(x, PHASES, SYNCF, SYNCBF);
    k_gemm_bias<<<2048, 128, 0, stream>>>(SYNCBF, SW1P, sb1, T1, 8192, 288, 256);
    k_ln_act<<<1024, 256, 0, stream>>>(T1, slng, slnb, H1BF, 8192, /*gelu*/1);
    k_gemm_bias<<<1024, 128, 0, stream>>>(H1BF, SW2P, sb2, T2, 8192, 256, 128);
    k_ew_act<<<1024, 256, 0, stream>>>(T2, 8192 * 128, /*gelu*/1);
    k_gemv128<<<1024, 256, 0, stream>>>(T2, sw3, sb3, out, 8192, 1, nullptr, nullptr);

    // --- order parameter / phase std / coherence + phases copy ---
    k_stats<<<1, 1024, 0, stream>>>(PHASES, out);
}
